// SimpleLSTM_46248207843581
// MI455X (gfx1250) — matrix operand order repaired and checked by reading, not compiled, not run on hardware
//
#include <hip/hip_runtime.h>
#include <stdint.h>

// ---------------------------------------------------------------------------
// Persistent-workgroup LSTM for MI455X (gfx1250, wave32, WMMA).
//   grid = 2 workgroups (16 batches each), block = 768 threads = 24 waves.
//   Per step: 24 waves x 3 x V_WMMA_F32_16X16X32_F16 -> gates[16x384],
//   branchless activation with exp-argument constant folding (v_exp_f32 is
//   base-2, so fold amul*log2e into the linear gate pre-activation),
//   c/h update + dense head, 2 barriers/step.
// ---------------------------------------------------------------------------

typedef __attribute__((ext_vector_type(16))) _Float16 v16h;
typedef __attribute__((ext_vector_type(8)))  float    v8f;
typedef unsigned int u32x4 __attribute__((ext_vector_type(4)));

#define HID    96
#define G4     384         // 4*H
#define BB     16          // batches per workgroup
#define T_LEN  16384
#define NWAVE  24
#define NTHR   (NWAVE * 32)
#define LOG2E  1.4426950408889634f

__device__ __forceinline__ float fast_tanh(float v) {
    // tanh(v) = 2 / (1 + 2^(-2*log2e*v)) - 1  (saturates at +/-1, no NaN)
    const float e = __builtin_amdgcn_exp2f(v * (-2.0f * LOG2E));
    return 2.0f * __builtin_amdgcn_rcpf(1.0f + e) - 1.0f;
}

__global__ __launch_bounds__(NTHR, 1)
void lstm_persistent_wmma(const float* __restrict__ x,      // [32, T]
                          const float* __restrict__ w_ih,   // [384]
                          const float* __restrict__ w_hh,   // [384, 96]
                          const float* __restrict__ b_ih,   // [384]
                          const float* __restrict__ b_hh,   // [384]
                          const float* __restrict__ w_dense,// [96]
                          const float* __restrict__ b_dense,// [1]
                          float* __restrict__ out)          // [32, T]
{
    // LDS state (per WGP, well under limits): ~21.5 KB
    __shared__ __align__(16) _Float16 h_lds[BB * HID];    // h  [b][k], f16 (A-matrix source)
    __shared__ __align__(16) _Float16 gates_lds[G4 * BB]; // activated gates [g][b], f16
    __shared__ __align__(16) float xs[BB];                // x column for current t
    __shared__ float c_lds[BB * HID];                     // c (job-indexed)
    __shared__ float y_lds[BB];                           // dense-head accumulators
    __shared__ float wd[HID];                             // w_dense cached

    const int tid  = threadIdx.x;
    const int lane = tid & 31;
    const int wv   = tid >> 5;           // wave id 0..23 -> N-tile (gate columns)
    const int b0   = blockIdx.x * BB;    // global batch base

    const int n    = lane & 15;          // column within tile / batch row for A
    const int hi   = lane >> 4;          // lane half selector
    const int g    = wv * 16 + n;        // this lane's gate column, 0..383
    const int gtyp = wv / 6;             // 0:i 1:f 2:g 3:o (uniform per wave)

    // Branchless activation: act(v) = bmul * rcp(1 + 2^(ea*v + folded)) + cadd
    //   sigmoid: amul=-1, bmul=1, cadd=0
    //   tanh:    amul=-2, bmul=2, cadd=-1   (tanh(v) = 2*sigmoid(2v)-1)
    const float amul = (gtyp == 2) ? -2.0f : -1.0f;
    const float bmul = (gtyp == 2) ?  2.0f :  1.0f;
    const float cadd = (gtyp == 2) ? -1.0f :  0.0f;

    const float xw   = w_ih[g];
    const float bias = b_ih[g] + b_hh[g];
    const float bd   = b_dense[0];

    // Fold amul and log2e into the linear pre-activation:
    //   exparg = ea*acc + x*(ea*xw) + ea*bias    -> 2 fmas per element
    const float ea  = amul * LOG2E;
    const float xw2 = ea * xw;
    const float bz  = ea * bias;

    // ---- build persistent B fragments (B[k][n] = w_hh[g][k]), f32 -> f16 ----
    // B 32x16 f16 order used here (contiguous K, same as A): lanes 0-15 hold K=0..15, lanes 16-31 K=16..31,
    // VGPR v holds the K pair (kbase + 2v, kbase + 2v + 1).
    v16h bf[3];
    #pragma unroll
    for (int kt = 0; kt < 3; ++kt) {
        union { v16h vec; _Float16 e[16]; } u;
        #pragma unroll
        for (int p = 0; p < 8; ++p) {
            const int k = kt * 32 + hi * 16 + 2 * p;
            u.e[2 * p]     = (_Float16)w_hh[g * HID + k];
            u.e[2 * p + 1] = (_Float16)w_hh[g * HID + k + 1];
        }
        bf[kt] = u.vec;
    }

    // ---- init state ----
    for (int i = tid; i < BB * HID; i += NTHR) {
        c_lds[i] = 0.0f;
        h_lds[i] = (_Float16)0.0f;
    }
    for (int i = tid; i < HID; i += NTHR) wd[i] = w_dense[i];
    if (tid < BB) {
        xs[tid]    = x[(b0 + tid) * T_LEN + 0];
        y_lds[tid] = 0.0f;
    }
    __syncthreads();

    for (int t = 0; t < T_LEN; ++t) {
        // ============ phase A: gates = h @ w_hh^T + x*w_ih + b, activated ====
        v8f acc = {0.f, 0.f, 0.f, 0.f, 0.f, 0.f, 0.f, 0.f};
        #pragma unroll
        for (int kt = 0; kt < 3; ++kt) {
            // A 16x32 f16 layout: lane half picks K 0..15 / 16..31 (matches B);
            // contiguous K -> two ds_load_b128 per tile.
            union { v16h vec; u32x4 q[2]; } a;
            const _Float16* base = &h_lds[n * HID + kt * 32 + hi * 16];
            a.q[0] = *(const u32x4*)(base);
            a.q[1] = *(const u32x4*)(base + 8);
            acc = __builtin_amdgcn_wmma_f32_16x16x32_f16(
                      false, a.vec, false, bf[kt], (short)0, acc, false, false);
        }

        // preload this lane's 8 x-values (contiguous) as two b128 LDS loads
        union { u32x4 q[2]; float f[8]; } xv;
        xv.q[0] = *(const u32x4*)(&xs[8 * hi]);
        xv.q[1] = *(const u32x4*)(&xs[8 * hi + 4]);

        // D layout: lane holds column g, element e is batch row (e + 8*hi).
        // 2 fmas -> v_exp_f32 -> v_rcp_f32 -> fma, then pack 8 f16, b128 store.
        union { u32x4 q; _Float16 e[8]; } pk;
        #pragma unroll
        for (int e = 0; e < 8; ++e) {
            const float exparg = fmaf(acc[e], ea, fmaf(xv.f[e], xw2, bz));
            const float r = __builtin_amdgcn_rcpf(1.0f + __builtin_amdgcn_exp2f(exparg));
            pk.e[e] = (_Float16)fmaf(bmul, r, cadd);
        }
        *(u32x4*)(&gates_lds[g * BB + 8 * hi]) = pk.q;
        __syncthreads();   // barrier 1: gates ready

        // ============ phase B: c/h update + head accumulation ===============
        #pragma unroll
        for (int j = 0; j < 2; ++j) {
            const int job = tid + j * NTHR;   // 0..1535
            const int b   = job & 15;
            const int k   = job >> 4;         // 0..95
            const float iv = (float)gates_lds[(k      ) * BB + b];
            const float fv = (float)gates_lds[(k +  96) * BB + b];
            const float gg = (float)gates_lds[(k + 192) * BB + b];
            const float ov = (float)gates_lds[(k + 288) * BB + b];
            const float c  = fv * c_lds[job] + iv * gg;
            c_lds[job] = c;
            const float hv = ov * fast_tanh(c);
            h_lds[b * HID + k] = (_Float16)hv;
            atomicAdd(&y_lds[b], hv * wd[k]);   // ds_add_f32 (dense head)
        }
        // stage next x column (xs reads of step t finished before barrier 1)
        if (tid < BB) {
            xs[tid] = (t + 1 < T_LEN) ? x[(b0 + tid) * T_LEN + t + 1] : 0.0f;
        }
        __syncthreads();   // barrier 2: h/c/y ready

        // head writeback + reset (safe: next adds occur only after barrier 1)
        if (tid < BB) {
            out[(b0 + tid) * T_LEN + t] = y_lds[tid] + bd;
            y_lds[tid] = 0.0f;
        }
    }
}

extern "C" void kernel_launch(void* const* d_in, const int* in_sizes, int n_in,
                              void* d_out, int out_size, void* d_ws, size_t ws_size,
                              hipStream_t stream) {
    const float* x    = (const float*)d_in[0];
    const float* wih  = (const float*)d_in[1];
    const float* whh  = (const float*)d_in[2];
    const float* bih  = (const float*)d_in[3];
    const float* bhh  = (const float*)d_in[4];
    const float* wdns = (const float*)d_in[5];
    const float* bdns = (const float*)d_in[6];
    float* out = (float*)d_out;

    hipLaunchKernelGGL(lstm_persistent_wmma, dim3(2), dim3(NTHR), 0, stream,
                       x, wih, whh, bih, bhh, wdns, bdns, out);
}
